// lps_real_62285615727213
// MI455X (gfx1250) — matrix operand order repaired and checked by reading, not compiled, not run on hardware
//
#include <hip/hip_runtime.h>
#include <hip/hip_bf16.h>

typedef __attribute__((ext_vector_type(16))) _Float16 v16h;
typedef __attribute__((ext_vector_type(8)))  float    v8f;
typedef _Float16 h16;

#define DEV __device__ __forceinline__

union V16 { v16h v; h16 e[16]; };
union V8F { v8f v; float e[8]; };

DEV v8f wmma16(v16h a, v16h b, v8f c) {
  // D = A(16x32 f16) * B(32x16 f16) + C(16x16 f32)
  return __builtin_amdgcn_wmma_f32_16x16x32_f16(false, a, false, b, (short)0, c, false, false);
}

DEV void wave_fence() { asm volatile("s_wait_dscnt 0x0" ::: "memory"); }

// Fast transcendental gates: raw v_rcp_f32 instead of the IEEE divide sequence
// (the GRU recurrence is a latency chain; this is on the critical path).
DEV float sigmoidf_(float x) { return __builtin_amdgcn_rcpf(1.0f + __expf(-x)); }
DEV float tanhf_(float x) { return 1.0f - 2.0f * __builtin_amdgcn_rcpf(__expf(2.0f * x) + 1.0f); }

DEV V8F bc8(float b) {
  V8F c;
  #pragma unroll
  for (int r = 0; r < 8; r++) c.e[r] = b;
  return c;
}

DEV V8F gru_gate(V8F gir, V8F giz, V8F gin, V8F ghr, V8F ghz, V8F ghn, V8F h) {
  V8F o;
  #pragma unroll
  for (int r = 0; r < 8; r++) {
    float rr = sigmoidf_(gir.e[r] + ghr.e[r]);
    float zz = sigmoidf_(giz.e[r] + ghz.e[r]);
    float nn = tanhf_(gin.e[r] + rr * ghn.e[r]);
    o.e[r] = (1.0f - zz) * nn + zz * h.e[r];
  }
  return o;
}

// ---------------------------------------------------------------------------
// inverse-filter tap tables: full_mask[f] = w0[f]*bm[k0[f]] + w1[f]*bm[k0[f]+1]
// ---------------------------------------------------------------------------
__global__ void k_tables(float* w0, float* w1, int* k0) {
  const int EB[32] = {0,3,5,8,10,13,15,18,22,25,29,33,38,42,48,53,59,66,73,80,
                      89,97,107,117,128,140,153,167,183,199,216,256};
  int f = blockIdx.x * blockDim.x + threadIdx.x;
  if (f > 256) return;
  float a = 0.f, bwt = 0.f; int k = 0;
  if (f < 256) {
    for (int i = 0; i < 31; i++) {
      if (f >= EB[i] && f < EB[i + 1]) {
        int w = EB[i + 1] - EB[i]; int j = f - EB[i];
        a = 1.0f - (float)j / (float)w; bwt = (float)j / (float)w; k = i;
      }
    }
  }
  w0[f] = a; w1[f] = bwt; k0[f] = k;
}

// ---------------------------------------------------------------------------
// Combine stage1 fc_in (64->32) with gru0 wih (32->96): Wc[96][64], bc[96]
// ---------------------------------------------------------------------------
__global__ void k_combine(const float* wih0, const float* fcw, const float* fcb,
                          const float* bih0, float* Wc, float* bc) {
  int n = blockIdx.x, k = threadIdx.x;       // 96 x 64
  float s = 0.f;
  for (int j = 0; j < 32; j++) s += wih0[n * 32 + j] * fcw[j * 64 + k];
  Wc[n * 64 + k] = s;
  if (k == 0) {
    float t = bih0[n];
    for (int j = 0; j < 32; j++) t += wih0[n * 32 + j] * fcb[j];
    bc[n] = t;
  }
}

// ---------------------------------------------------------------------------
// Pack f32 weight into gfx1250 WMMA B-operand layout (f16):
//   dst[((nt*KC+kc)*32 + lane)*16 + i] = W(k = kc*32 + (i<8 ? 8*(lane>=16) + i : 16 + 8*(lane>=16) + i-8),
//                                          n = nt*16 + (lane&15))
// ---------------------------------------------------------------------------
__global__ void k_pack(const float* src, h16* dst, int Kv, int Nv, int ld,
                       int trans, int KC, int NT) {
  int idx = blockIdx.x * blockDim.x + threadIdx.x;
  int total = NT * KC * 512;
  if (idx >= total) return;
  int i = idx & 15, lane = (idx >> 4) & 31, tile = idx >> 9;
  int kc = tile % KC, nt = tile / KC;
  int k = kc * 32 + ((i < 8) ? (((lane & 16) ? 8 : 0) + i) : (16 + ((lane & 16) ? 8 : 0) + (i - 8)));
  int n = nt * 16 + (lane & 15);
  float v = 0.f;
  if (k < Kv && n < Nv) v = trans ? src[n * ld + k] : src[k * ld + n];
  dst[idx] = (h16)v;
}

// ---------------------------------------------------------------------------
// Stage1 sequential: fused GRU0+GRU1+fcout over T; one wave = 16 batch rows.
// ---------------------------------------------------------------------------
__global__ __launch_bounds__(32, 1) void k_seq_stage1(
    const float* __restrict__ mainin, const h16* pb_gi, const h16* pb_whh0,
    const h16* pb_wih1, const h16* pb_whh1, const h16* pb_fco,
    const float* bcv, const float* bhh0, const float* bih1, const float* bhh1,
    float* bm32, h16* bm16) {
  __shared__ h16 lds[16 * 32];
  int lane = threadIdx.x;
  int b0 = blockIdx.x * 16;
  int lm = lane & 15;
  int hi = (lane & 16) ? 1 : 0;

  V16 Wgi[12], W0[6], W1i[6], W1h[6], Wf[2];
  #pragma unroll
  for (int i = 0; i < 12; i++) Wgi[i].v = *(const v16h*)(pb_gi + (i * 32 + lane) * 16);
  #pragma unroll
  for (int i = 0; i < 6; i++) {
    W0[i].v  = *(const v16h*)(pb_whh0 + (i * 32 + lane) * 16);
    W1i[i].v = *(const v16h*)(pb_wih1 + (i * 32 + lane) * 16);
    W1h[i].v = *(const v16h*)(pb_whh1 + (i * 32 + lane) * 16);
  }
  Wf[0].v = *(const v16h*)(pb_fco + (0 * 32 + lane) * 16);
  Wf[1].v = *(const v16h*)(pb_fco + (1 * 32 + lane) * 16);

  float bgi[6], bh0[6], bi1[6], bh1[6];
  #pragma unroll
  for (int nt = 0; nt < 6; nt++) {
    bgi[nt] = bcv[nt * 16 + lm]; bh0[nt] = bhh0[nt * 16 + lm];
    bi1[nt] = bih1[nt * 16 + lm]; bh1[nt] = bhh1[nt * 16 + lm];
  }

  V8F h0c[2], h1c[2];
  #pragma unroll
  for (int ct = 0; ct < 2; ct++) { h0c[ct] = bc8(0.f); h1c[ct] = bc8(0.f); }
  V16 h0a, h1a;
  #pragma unroll
  for (int i = 0; i < 16; i++) { h0a.e[i] = (h16)0.f; h1a.e[i] = (h16)0.f; }

  const float* mrow = mainin + (long)(b0 + lm) * 256 * 32;

  for (int t = 0; t < 256; t++) {
    if (t < 255) __builtin_prefetch(mrow + (t + 1) * 32, 0, 0);
    // --- A operand for gi0 (windowed input, K=64), on the fly ---
    V16 A0[2];
    #pragma unroll
    for (int kc = 0; kc < 2; kc++) {
      int K0 = kc * 32 + hi * 8;
      #pragma unroll
      for (int j = 0; j < 8; j++) {
        int K = K0 + j; int tt = t + (K & 1) - 1;
        A0[kc].e[j] = (h16)((tt >= 0) ? mrow[tt * 32 + (K >> 1)] : 0.f);
        K = K0 + 16 + j; tt = t + (K & 1) - 1;
        A0[kc].e[j + 8] = (h16)((tt >= 0) ? mrow[tt * 32 + (K >> 1)] : 0.f);
      }
    }
    // --- GRU0 gates ---
    V8F gi0[6], gh0[6];
    #pragma unroll
    for (int nt = 0; nt < 6; nt++) {
      V8F c = bc8(bgi[nt]);
      c.v = wmma16(A0[0].v, Wgi[nt * 2 + 0].v, c.v);
      c.v = wmma16(A0[1].v, Wgi[nt * 2 + 1].v, c.v);
      gi0[nt] = c;
      V8F d = bc8(bh0[nt]);
      d.v = wmma16(h0a.v, W0[nt].v, d.v);
      gh0[nt] = d;
    }
    #pragma unroll
    for (int ct = 0; ct < 2; ct++)
      h0c[ct] = gru_gate(gi0[ct], gi0[ct + 2], gi0[ct + 4],
                         gh0[ct], gh0[ct + 2], gh0[ct + 4], h0c[ct]);
    // --- h0 (== y1) C-layout -> A-layout via LDS ---
    #pragma unroll
    for (int ct = 0; ct < 2; ct++)
      #pragma unroll
      for (int r = 0; r < 8; r++)
        lds[(r + hi * 8) * 32 + ct * 16 + lm] = (h16)h0c[ct].e[r];
    wave_fence();
    { int base = lm * 32 + hi * 8;
      #pragma unroll
      for (int i = 0; i < 8; i++) { h0a.e[i] = lds[base + i]; h0a.e[i + 8] = lds[base + 16 + i]; } }
    wave_fence();
    // --- GRU1 ---
    V8F gi1[6], gh1[6];
    #pragma unroll
    for (int nt = 0; nt < 6; nt++) {
      V8F c = bc8(bi1[nt]); c.v = wmma16(h0a.v, W1i[nt].v, c.v); gi1[nt] = c;
      V8F d = bc8(bh1[nt]); d.v = wmma16(h1a.v, W1h[nt].v, d.v); gh1[nt] = d;
    }
    #pragma unroll
    for (int ct = 0; ct < 2; ct++)
      h1c[ct] = gru_gate(gi1[ct], gi1[ct + 2], gi1[ct + 4],
                         gh1[ct], gh1[ct + 2], gh1[ct + 4], h1c[ct]);
    // --- x1 = y1 + y2 -> A via LDS ---
    #pragma unroll
    for (int ct = 0; ct < 2; ct++)
      #pragma unroll
      for (int r = 0; r < 8; r++)
        lds[(r + hi * 8) * 32 + ct * 16 + lm] = (h16)(h0c[ct].e[r] + h1c[ct].e[r]);
    wave_fence();
    V16 x1a;
    { int base = lm * 32 + hi * 8;
      #pragma unroll
      for (int i = 0; i < 8; i++) { x1a.e[i] = lds[base + i]; x1a.e[i + 8] = lds[base + 16 + i]; } }
    wave_fence();
    // --- h1 -> A for next step ---
    #pragma unroll
    for (int ct = 0; ct < 2; ct++)
      #pragma unroll
      for (int r = 0; r < 8; r++)
        lds[(r + hi * 8) * 32 + ct * 16 + lm] = (h16)h1c[ct].e[r];
    wave_fence();
    { int base = lm * 32 + hi * 8;
      #pragma unroll
      for (int i = 0; i < 8; i++) { h1a.e[i] = lds[base + i]; h1a.e[i + 8] = lds[base + 16 + i]; } }
    wave_fence();
    // --- band_mask = sigmoid(x1 @ fcout) ---
    #pragma unroll
    for (int ct = 0; ct < 2; ct++) {
      V8F d = bc8(0.f);
      d.v = wmma16(x1a.v, Wf[ct].v, d.v);
      #pragma unroll
      for (int r = 0; r < 8; r++) {
        float s = sigmoidf_(d.e[r]);
        long o = ((long)(b0 + r + hi * 8) * 256 + t) * 32 + ct * 16 + lm;
        bm32[o] = s; bm16[o] = (h16)s;
      }
    }
  }
}

// ---------------------------------------------------------------------------
// Low-band input projection: y = prelu([logp[:,:128], bm] @ W160x64); f16 out.
// ---------------------------------------------------------------------------
__global__ __launch_bounds__(128, 1) void k_low_in(
    const float* __restrict__ logp, const h16* __restrict__ bm16,
    const h16* __restrict__ pb, const float* prelu, h16* __restrict__ y16) {
  int lane = threadIdx.x & 31;
  int tile = blockIdx.x * 4 + (threadIdx.x >> 5);
  int flat0 = tile * 16;
  int b = flat0 >> 8, t0 = flat0 & 255;
  int lm = lane & 15, hi = (lane & 16) ? 1 : 0;
  float al = prelu[0];
  long rowi = (long)(b * 256 + t0 + lm);
  V16 A[5];
  #pragma unroll
  for (int kc = 0; kc < 4; kc++) {
    const float* p = logp + rowi * 256 + kc * 32 + hi * 8;
    #pragma unroll
    for (int i = 0; i < 8; i++) { A[kc].e[i] = (h16)p[i]; A[kc].e[i + 8] = (h16)p[i + 16]; }
  }
  { const h16* p = bm16 + rowi * 32 + hi * 8;
    #pragma unroll
    for (int i = 0; i < 8; i++) { A[4].e[i] = p[i]; A[4].e[i + 8] = p[i + 16]; } }
  #pragma unroll
  for (int nt = 0; nt < 4; nt++) {
    V8F c = bc8(0.f);
    #pragma unroll
    for (int kc = 0; kc < 5; kc++) {
      V16 w; w.v = *(const v16h*)(pb + ((nt * 5 + kc) * 32 + lane) * 16);
      c.v = wmma16(A[kc].v, w.v, c.v);
    }
    #pragma unroll
    for (int r = 0; r < 8; r++) {
      float v = c.e[r]; v = (v >= 0.f) ? v : al * v;
      y16[(long)(flat0 + r + hi * 8) * 64 + nt * 16 + lm] = (h16)v;
    }
  }
}

// ---------------------------------------------------------------------------
// High-band input projection (grouped 80->16, PReLU); f16 out [B*T,32].
// ---------------------------------------------------------------------------
__global__ __launch_bounds__(128, 1) void k_high_in(
    const float* __restrict__ logp, const h16* __restrict__ bm16,
    const h16* __restrict__ pb, const float* prelu, h16* __restrict__ xh16) {
  int lane = threadIdx.x & 31;
  int tile = blockIdx.x * 4 + (threadIdx.x >> 5);
  int flat0 = tile * 16;
  int b = flat0 >> 8, t0 = flat0 & 255;
  int lm = lane & 15, hi = (lane & 16) ? 1 : 0;
  float ah = prelu[0];
  long rowi = (long)(b * 256 + t0 + lm);
  #pragma unroll
  for (int g = 0; g < 2; g++) {
    V16 A[3];
    if (g == 0) {
      #pragma unroll
      for (int kc = 0; kc < 2; kc++) {
        const float* p = logp + rowi * 256 + 128 + kc * 32 + hi * 8;
        #pragma unroll
        for (int i = 0; i < 8; i++) { A[kc].e[i] = (h16)p[i]; A[kc].e[i + 8] = (h16)p[i + 16]; }
      }
      { const float* p = logp + rowi * 256 + 192 + hi * 8;
        #pragma unroll
        for (int i = 0; i < 8; i++) { A[2].e[i] = (h16)p[i]; A[2].e[i + 8] = (h16)0.f; } }
    } else {
      { const float* p = logp + rowi * 256 + 208 + hi * 8;
        #pragma unroll
        for (int i = 0; i < 8; i++) { A[0].e[i] = (h16)p[i]; A[0].e[i + 8] = (h16)p[i + 16]; } }
      { const float* p1 = logp + rowi * 256 + 240 + hi * 8;
        const h16* p2 = bm16 + rowi * 32 + hi * 8;
        #pragma unroll
        for (int i = 0; i < 8; i++) { A[1].e[i] = (h16)p1[i]; A[1].e[i + 8] = p2[i]; } }
      { const h16* p = bm16 + rowi * 32 + 16 + hi * 8;
        #pragma unroll
        for (int i = 0; i < 8; i++) { A[2].e[i] = p[i]; A[2].e[i + 8] = (h16)0.f; } }
    }
    V8F c = bc8(0.f);
    #pragma unroll
    for (int kc = 0; kc < 3; kc++) {
      V16 w; w.v = *(const v16h*)(pb + ((g * 3 + kc) * 32 + lane) * 16);
      c.v = wmma16(A[kc].v, w.v, c.v);
    }
    #pragma unroll
    for (int r = 0; r < 8; r++) {
      float v = c.e[r]; v = (v >= 0.f) ? v : ah * v;
      xh16[(long)(flat0 + r + hi * 8) * 32 + g * 16 + lm] = (h16)v;
    }
  }
}

// ---------------------------------------------------------------------------
// Low-band sequential: grouped GRU (2 layers, 2 groups, H=32), shuffle, acc.
// ---------------------------------------------------------------------------
__global__ __launch_bounds__(32, 1) void k_seq_low(
    const h16* __restrict__ y16, const h16* pb_i0, const h16* pb_h0,
    const h16* pb_i1, const h16* pb_h1,
    const float* bih, const float* bhh, h16* __restrict__ acc16) {
  __shared__ h16 ldsS[16 * 64];
  __shared__ h16 ldsC[16 * 64];
  int lane = threadIdx.x;
  int b0 = blockIdx.x * 16;
  int lm = lane & 15, hi = (lane & 16) ? 1 : 0;

  V16 Wi0[2][6], Wh0[2][6], Wi1[2][6], Wh1[2][6];
  #pragma unroll
  for (int g = 0; g < 2; g++)
    #pragma unroll
    for (int nt = 0; nt < 6; nt++) {
      Wi0[g][nt].v = *(const v16h*)(pb_i0 + ((g * 6 + nt) * 32 + lane) * 16);
      Wh0[g][nt].v = *(const v16h*)(pb_h0 + ((g * 6 + nt) * 32 + lane) * 16);
      Wi1[g][nt].v = *(const v16h*)(pb_i1 + ((g * 6 + nt) * 32 + lane) * 16);
      Wh1[g][nt].v = *(const v16h*)(pb_h1 + ((g * 6 + nt) * 32 + lane) * 16);
    }
  float bi0[2][6], bh0b[2][6], bi1[2][6], bh1b[2][6];
  #pragma unroll
  for (int g = 0; g < 2; g++)
    #pragma unroll
    for (int nt = 0; nt < 6; nt++) {
      bi0[g][nt]  = bih[(0 * 2 + g) * 96 + nt * 16 + lm];
      bh0b[g][nt] = bhh[(0 * 2 + g) * 96 + nt * 16 + lm];
      bi1[g][nt]  = bih[(1 * 2 + g) * 96 + nt * 16 + lm];
      bh1b[g][nt] = bhh[(1 * 2 + g) * 96 + nt * 16 + lm];
    }

  V8F h0c[2][2], h1c[2][2];
  #pragma unroll
  for (int g = 0; g < 2; g++)
    #pragma unroll
    for (int ct = 0; ct < 2; ct++) { h0c[g][ct] = bc8(0.f); h1c[g][ct] = bc8(0.f); }
  V16 h0a[2], h1a[2];
  #pragma unroll
  for (int g = 0; g < 2; g++)
    #pragma unroll
    for (int i = 0; i < 16; i++) { h0a[g].e[i] = (h16)0.f; h1a[g].e[i] = (h16)0.f; }

  for (int t = 0; t < 256; t++) {
    if (t < 255) __builtin_prefetch(y16 + ((long)(b0 + lm) * 256 + t + 1) * 64, 0, 0);
    // --- layer 0 (per group), gi from precomputed y16 ---
    #pragma unroll
    for (int g = 0; g < 2; g++) {
      V16 Ay;
      const h16* p = y16 + ((long)(b0 + lm) * 256 + t) * 64 + g * 32 + hi * 8;
      #pragma unroll
      for (int i = 0; i < 8; i++) { Ay.e[i] = p[i]; Ay.e[i + 8] = p[i + 16]; }
      V8F gi[6], gh[6];
      #pragma unroll
      for (int nt = 0; nt < 6; nt++) {
        V8F c = bc8(bi0[g][nt]); c.v = wmma16(Ay.v, Wi0[g][nt].v, c.v); gi[nt] = c;
        V8F d = bc8(bh0b[g][nt]); d.v = wmma16(h0a[g].v, Wh0[g][nt].v, d.v); gh[nt] = d;
      }
      #pragma unroll
      for (int ct = 0; ct < 2; ct++)
        h0c[g][ct] = gru_gate(gi[ct], gi[ct + 2], gi[ct + 4],
                              gh[ct], gh[ct + 2], gh[ct + 4], h0c[g][ct]);
    }
    // --- channel shuffle of layer0 output into LDS ---
    #pragma unroll
    for (int g = 0; g < 2; g++)
      #pragma unroll
      for (int ct = 0; ct < 2; ct++)
        #pragma unroll
        for (int r = 0; r < 8; r++) {
          int cch = g * 32 + ct * 16 + lm;
          int k = (cch & 1) ? (32 + (cch >> 1)) : (cch >> 1);
          ldsS[(r + hi * 8) * 64 + k] = (h16)h0c[g][ct].e[r];
        }
    wave_fence();
    // --- layer 1 ---
    V16 As[2];
    #pragma unroll
    for (int g = 0; g < 2; g++) {
      int base = lm * 64 + g * 32 + hi * 8;
      #pragma unroll
      for (int i = 0; i < 8; i++) { As[g].e[i] = ldsS[base + i]; As[g].e[i + 8] = ldsS[base + 16 + i]; }
    }
    #pragma unroll
    for (int g = 0; g < 2; g++) {
      V8F gi[6], gh[6];
      #pragma unroll
      for (int nt = 0; nt < 6; nt++) {
        V8F c = bc8(bi1[g][nt]); c.v = wmma16(As[g].v, Wi1[g][nt].v, c.v); gi[nt] = c;
        V8F d = bc8(bh1b[g][nt]); d.v = wmma16(h1a[g].v, Wh1[g][nt].v, d.v); gh[nt] = d;
      }
      #pragma unroll
      for (int ct = 0; ct < 2; ct++)
        h1c[g][ct] = gru_gate(gi[ct], gi[ct + 2], gi[ct + 4],
                              gh[ct], gh[ct + 2], gh[ct + 4], h1c[g][ct]);
    }
    // --- h0 -> A for next step ---
    #pragma unroll
    for (int g = 0; g < 2; g++)
      #pragma unroll
      for (int ct = 0; ct < 2; ct++)
        #pragma unroll
        for (int r = 0; r < 8; r++)
          ldsC[(r + hi * 8) * 64 + g * 32 + ct * 16 + lm] = (h16)h0c[g][ct].e[r];
    wave_fence();
    #pragma unroll
    for (int g = 0; g < 2; g++) {
      int base = lm * 64 + g * 32 + hi * 8;
      #pragma unroll
      for (int i = 0; i < 8; i++) { h0a[g].e[i] = ldsC[base + i]; h0a[g].e[i + 8] = ldsC[base + 16 + i]; }
    }
    // --- acc = shuffle(y0) + y1 -> global f16 ---
    #pragma unroll
    for (int g = 0; g < 2; g++)
      #pragma unroll
      for (int ct = 0; ct < 2; ct++)
        #pragma unroll
        for (int r = 0; r < 8; r++) {
          int gk = g * 32 + ct * 16 + lm;
          int m = r + hi * 8;
          float v = (float)ldsS[m * 64 + gk] + h1c[g][ct].e[r];
          acc16[((long)(b0 + m) * 256 + t) * 64 + gk] = (h16)v;
        }
    wave_fence();
    // --- h1 -> A for next step ---
    #pragma unroll
    for (int g = 0; g < 2; g++)
      #pragma unroll
      for (int ct = 0; ct < 2; ct++)
        #pragma unroll
        for (int r = 0; r < 8; r++)
          ldsC[(r + hi * 8) * 64 + g * 32 + ct * 16 + lm] = (h16)h1c[g][ct].e[r];
    wave_fence();
    #pragma unroll
    for (int g = 0; g < 2; g++) {
      int base = lm * 64 + g * 32 + hi * 8;
      #pragma unroll
      for (int i = 0; i < 8; i++) { h1a[g].e[i] = ldsC[base + i]; h1a[g].e[i + 8] = ldsC[base + 16 + i]; }
    }
    wave_fence();
  }
}

// ---------------------------------------------------------------------------
// High-band sequential: grouped GRU (2 groups, H=16, K padded to 32).
// ---------------------------------------------------------------------------
__global__ __launch_bounds__(32, 1) void k_seq_high(
    const h16* __restrict__ xh16, const h16* pb_i, const h16* pb_h,
    const float* bih, const float* bhh, h16* __restrict__ dfh16) {
  __shared__ h16 ldsH[2][16 * 16];
  int lane = threadIdx.x;
  int b0 = blockIdx.x * 16;
  int lm = lane & 15, hi = (lane & 16) ? 1 : 0;
  V16 Wi[2][3], Wh[2][3];
  #pragma unroll
  for (int g = 0; g < 2; g++)
    #pragma unroll
    for (int nt = 0; nt < 3; nt++) {
      Wi[g][nt].v = *(const v16h*)(pb_i + ((g * 3 + nt) * 32 + lane) * 16);
      Wh[g][nt].v = *(const v16h*)(pb_h + ((g * 3 + nt) * 32 + lane) * 16);
    }
  float bi[2][3], bh[2][3];
  #pragma unroll
  for (int g = 0; g < 2; g++)
    #pragma unroll
    for (int nt = 0; nt < 3; nt++) {
      bi[g][nt] = bih[g * 48 + nt * 16 + lm];
      bh[g][nt] = bhh[g * 48 + nt * 16 + lm];
    }
  V8F hc[2]; hc[0] = bc8(0.f); hc[1] = bc8(0.f);
  V16 ha[2];
  #pragma unroll
  for (int g = 0; g < 2; g++)
    #pragma unroll
    for (int i = 0; i < 16; i++) ha[g].e[i] = (h16)0.f;

  for (int t = 0; t < 256; t++) {
    #pragma unroll
    for (int g = 0; g < 2; g++) {
      V16 Ax;
      const h16* p = xh16 + ((long)(b0 + lm) * 256 + t) * 32 + g * 16 + hi * 8;
      #pragma unroll
      for (int i = 0; i < 8; i++) { Ax.e[i] = p[i]; Ax.e[i + 8] = (h16)0.f; }
      V8F gi[3], gh[3];
      #pragma unroll
      for (int nt = 0; nt < 3; nt++) {
        V8F c = bc8(bi[g][nt]); c.v = wmma16(Ax.v, Wi[g][nt].v, c.v); gi[nt] = c;
        V8F d = bc8(bh[g][nt]); d.v = wmma16(ha[g].v, Wh[g][nt].v, d.v); gh[nt] = d;
      }
      hc[g] = gru_gate(gi[0], gi[1], gi[2], gh[0], gh[1], gh[2], hc[g]);
      #pragma unroll
      for (int r = 0; r < 8; r++) {
        int m = r + hi * 8;
        dfh16[((long)(b0 + m) * 256 + t) * 32 + g * 16 + lm] = (h16)hc[g].e[r];
        ldsH[g][m * 16 + lm] = (h16)hc[g].e[r];
      }
    }
    wave_fence();
    #pragma unroll
    for (int g = 0; g < 2; g++) {
      int base = lm * 16 + hi * 8;
      #pragma unroll
      for (int i = 0; i < 8; i++) { ha[g].e[i] = ldsH[g][base + i]; ha[g].e[i + 8] = (h16)0.f; }
    }
    wave_fence();
  }
}

// ---------------------------------------------------------------------------
// mask_low = sigmoid(acc @ W64x128)
// ---------------------------------------------------------------------------
__global__ __launch_bounds__(128, 1) void k_fcout_low(
    const h16* __restrict__ acc16, const h16* __restrict__ pb, float* __restrict__ mlow) {
  int lane = threadIdx.x & 31;
  int tile = blockIdx.x * 4 + (threadIdx.x >> 5);
  int flat0 = tile * 16;
  int lm = lane & 15, hi = (lane & 16) ? 1 : 0;
  V16 A[2];
  const h16* arow = acc16 + (long)(flat0 + lm) * 64;
  #pragma unroll
  for (int kc = 0; kc < 2; kc++) {
    const h16* p = arow + kc * 32 + hi * 8;
    #pragma unroll
    for (int i = 0; i < 8; i++) { A[kc].e[i] = p[i]; A[kc].e[i + 8] = p[i + 16]; }
  }
  #pragma unroll
  for (int nt = 0; nt < 8; nt++) {
    V8F c = bc8(0.f);
    #pragma unroll
    for (int kc = 0; kc < 2; kc++) {
      V16 w; w.v = *(const v16h*)(pb + ((nt * 2 + kc) * 32 + lane) * 16);
      c.v = wmma16(A[kc].v, w.v, c.v);
    }
    #pragma unroll
    for (int r = 0; r < 8; r++)
      mlow[(long)(flat0 + r + hi * 8) * 128 + nt * 16 + lm] = sigmoidf_(c.e[r]);
  }
}

// ---------------------------------------------------------------------------
// mask_high = sigmoid(dfh @ W16x64 per group)
// ---------------------------------------------------------------------------
__global__ __launch_bounds__(128, 1) void k_fcout_high(
    const h16* __restrict__ dfh16, const h16* __restrict__ pb, float* __restrict__ mhigh) {
  int lane = threadIdx.x & 31;
  int tile = blockIdx.x * 4 + (threadIdx.x >> 5);
  int flat0 = tile * 16;
  int lm = lane & 15, hi = (lane & 16) ? 1 : 0;
  #pragma unroll
  for (int g = 0; g < 2; g++) {
    V16 A;
    const h16* p = dfh16 + (long)(flat0 + lm) * 32 + g * 16 + hi * 8;
    #pragma unroll
    for (int i = 0; i < 8; i++) { A.e[i] = p[i]; A.e[i + 8] = (h16)0.f; }
    #pragma unroll
    for (int nt = 0; nt < 4; nt++) {
      V8F c = bc8(0.f);
      V16 w; w.v = *(const v16h*)(pb + ((g * 4 + nt) * 32 + lane) * 16);
      c.v = wmma16(A.v, w.v, c.v);
      #pragma unroll
      for (int r = 0; r < 8; r++)
        mhigh[(long)(flat0 + r + hi * 8) * 128 + g * 64 + nt * 16 + lm] = sigmoidf_(c.e[r]);
    }
  }
}

// ---------------------------------------------------------------------------
// Fused epilogue: full_mask (2-tap fbank), mask fuse, spec_out, spec_mag.
// ---------------------------------------------------------------------------
__global__ void k_epilogue(const float* __restrict__ spec, const float* __restrict__ bm32,
                           const float* __restrict__ mlow, const float* __restrict__ mhigh,
                           const float* __restrict__ w0t, const float* __restrict__ w1t,
                           const int* __restrict__ k0t, float* __restrict__ out) {
  long idx = (long)blockIdx.x * 256 + threadIdx.x;
  const long N = 256L * 257L * 256L;
  if (idx >= N) return;
  int t = (int)(idx & 255);
  long r2 = idx >> 8;
  int f = (int)(r2 % 257);
  int b = (int)(r2 / 257);
  long bt = (long)(b * 256 + t);
  int k0 = k0t[f];
  int k1 = (k0 + 1 < 32) ? k0 + 1 : 31;
  float fm = w0t[f] * bm32[bt * 32 + k0] + w1t[f] * bm32[bt * 32 + k1];
  float sel = (f < 128) ? mlow[bt * 128 + f]
                        : ((f < 256) ? mhigh[bt * 128 + (f - 128)] : 1.0f);
  float m = sel * fm;
  float re = spec[idx * 2], im = spec[idx * 2 + 1];
  float sr = m * fm * re, si = m * fm * im;
  out[idx] = fm;
  out[N + idx] = m;
  out[2 * N + idx] = sqrtf(sr * sr + si * si);
  out[3 * N + idx * 2] = sr;
  out[3 * N + idx * 2 + 1] = si;
}

// ---------------------------------------------------------------------------
extern "C" void kernel_launch(void* const* d_in, const int* in_sizes, int n_in,
                              void* d_out, int out_size, void* d_ws, size_t ws_size,
                              hipStream_t stream) {
  const float* mainin = (const float*)d_in[0];
  const float* spec   = (const float*)d_in[1];
  const float* logp   = (const float*)d_in[2];
  const float* fcw    = (const float*)d_in[5];
  const float* fcb    = (const float*)d_in[6];
  const float* wih0   = (const float*)d_in[7];
  const float* whh0   = (const float*)d_in[8];
  const float* bih0   = (const float*)d_in[9];
  const float* bhh0   = (const float*)d_in[10];
  const float* wih1   = (const float*)d_in[11];
  const float* whh1   = (const float*)d_in[12];
  const float* bih1   = (const float*)d_in[13];
  const float* bhh1   = (const float*)d_in[14];
  const float* fco    = (const float*)d_in[15];
  const float* fcl    = (const float*)d_in[16];
  const float* prl    = (const float*)d_in[17];
  const float* fch    = (const float*)d_in[18];
  const float* prh    = (const float*)d_in[19];
  const float* glwih  = (const float*)d_in[20];
  const float* glwhh  = (const float*)d_in[21];
  const float* glbih  = (const float*)d_in[22];
  const float* glbhh  = (const float*)d_in[23];
  const float* ghwih  = (const float*)d_in[24];
  const float* ghwhh  = (const float*)d_in[25];
  const float* ghbih  = (const float*)d_in[26];
  const float* ghbhh  = (const float*)d_in[27];
  const float* fcol   = (const float*)d_in[28];
  const float* fcoh   = (const float*)d_in[29];

  char* wp = (char*)d_ws;
  auto alloc = [&](size_t bytes) -> char* {
    char* p = wp; wp += (bytes + 255) & ~(size_t)255; return p;
  };
  float* tw0   = (float*)alloc(257 * 4);
  float* tw1   = (float*)alloc(257 * 4);
  int*   tk0   = (int*)alloc(257 * 4);
  float* Wc    = (float*)alloc(96 * 64 * 4);
  float* bcv   = (float*)alloc(96 * 4);
  h16* pb_s1gi = (h16*)alloc(12 * 512 * 2);
  h16* pb_whh0 = (h16*)alloc(6 * 512 * 2);
  h16* pb_wih1 = (h16*)alloc(6 * 512 * 2);
  h16* pb_whh1 = (h16*)alloc(6 * 512 * 2);
  h16* pb_fco  = (h16*)alloc(2 * 512 * 2);
  h16* pb_fcl  = (h16*)alloc(20 * 512 * 2);
  h16* pb_i0   = (h16*)alloc(12 * 512 * 2);
  h16* pb_h0   = (h16*)alloc(12 * 512 * 2);
  h16* pb_i1   = (h16*)alloc(12 * 512 * 2);
  h16* pb_h1   = (h16*)alloc(12 * 512 * 2);
  h16* pb_fcol = (h16*)alloc(16 * 512 * 2);
  h16* pb_fch  = (h16*)alloc(6 * 512 * 2);
  h16* pb_hi   = (h16*)alloc(6 * 512 * 2);
  h16* pb_hh   = (h16*)alloc(6 * 512 * 2);
  h16* pb_fcoh = (h16*)alloc(8 * 512 * 2);
  h16*   y16   = (h16*)alloc(65536L * 64 * 2);
  h16*   xh16  = (h16*)alloc(65536L * 32 * 2);
  float* bm32  = (float*)alloc(65536L * 32 * 4);
  h16*   bm16  = (h16*)alloc(65536L * 32 * 2);
  h16*   acc16 = (h16*)alloc(65536L * 64 * 2);
  h16*   dfh16 = (h16*)alloc(65536L * 32 * 2);
  float* mlow  = (float*)alloc(65536L * 128 * 4);
  float* mhigh = (float*)alloc(65536L * 128 * 4);

  auto pack = [&](const float* src, h16* dst, int Kv, int Nv, int ld, int tr,
                  int KC, int NT) {
    int total = NT * KC * 512;
    k_pack<<<(total + 255) / 256, 256, 0, stream>>>(src, dst, Kv, Nv, ld, tr, KC, NT);
  };

  k_tables<<<2, 160, 0, stream>>>(tw0, tw1, tk0);
  k_combine<<<96, 64, 0, stream>>>(wih0, fcw, fcb, bih0, Wc, bcv);

  pack(Wc,   pb_s1gi, 64, 96, 64, 1, 2, 6);
  pack(whh0, pb_whh0, 32, 96, 32, 1, 1, 6);
  pack(wih1, pb_wih1, 32, 96, 32, 1, 1, 6);
  pack(whh1, pb_whh1, 32, 96, 32, 1, 1, 6);
  pack(fco,  pb_fco,  32, 32, 32, 0, 1, 2);
  pack(fcl,  pb_fcl, 160, 64, 64, 0, 5, 4);
  for (int g = 0; g < 2; g++) {
    pack(glwih + (0 * 2 + g) * 96 * 32, pb_i0 + g * 6 * 512, 32, 96, 32, 1, 1, 6);
    pack(glwhh + (0 * 2 + g) * 96 * 32, pb_h0 + g * 6 * 512, 32, 96, 32, 1, 1, 6);
    pack(glwih + (1 * 2 + g) * 96 * 32, pb_i1 + g * 6 * 512, 32, 96, 32, 1, 1, 6);
    pack(glwhh + (1 * 2 + g) * 96 * 32, pb_h1 + g * 6 * 512, 32, 96, 32, 1, 1, 6);
    pack(fch   + g * 80 * 16, pb_fch  + g * 3 * 512, 80, 16, 16, 0, 3, 1);
    pack(ghwih + g * 48 * 16, pb_hi   + g * 3 * 512, 16, 48, 16, 1, 1, 3);
    pack(ghwhh + g * 48 * 16, pb_hh   + g * 3 * 512, 16, 48, 16, 1, 1, 3);
    pack(fcoh  + g * 16 * 64, pb_fcoh + g * 4 * 512, 16, 64, 64, 0, 1, 4);
  }
  pack(fcol, pb_fcol, 64, 128, 128, 0, 2, 8);

  // Stage1 sequential (produces band_mask in f32 + f16)
  k_seq_stage1<<<16, 32, 0, stream>>>(mainin, pb_s1gi, pb_whh0, pb_wih1, pb_whh1,
                                      pb_fco, bcv, bhh0, bih1, bhh1, bm32, bm16);
  // Time-parallel input projections
  k_low_in<<<1024, 128, 0, stream>>>(logp, bm16, pb_fcl, prl, y16);
  k_high_in<<<1024, 128, 0, stream>>>(logp, bm16, pb_fch, prh, xh16);
  // Sequential band GRUs
  k_seq_low<<<16, 32, 0, stream>>>(y16, pb_i0, pb_h0, pb_i1, pb_h1, glbih, glbhh, acc16);
  k_seq_high<<<16, 32, 0, stream>>>(xh16, pb_hi, pb_hh, ghbih, ghbhh, dfh16);
  // Output projections
  k_fcout_low<<<1024, 128, 0, stream>>>(acc16, pb_fcol, mlow);
  k_fcout_high<<<1024, 128, 0, stream>>>(dfh16, pb_fcoh, mhigh);
  // Fused streaming epilogue
  long N = 256L * 257L * 256L;
  k_epilogue<<<(unsigned)((N + 255) / 256), 256, 0, stream>>>(
      spec, bm32, mlow, mhigh, tw0, tw1, tk0, (float*)d_out);
}
